// MultiHeadAttentionLayer_30580167148121
// MI455X (gfx1250) — matrix operand order repaired and checked by reading, not compiled, not run on hardware
//
#include <hip/hip_runtime.h>

// ---------------------------------------------------------------------------
// MI455X (gfx1250) wave32 bf16-WMMA implementation of the reference MHA.
// Pipeline:
//   [GEMM Q][GEMM K][GEMM V^T (transposed)] -> [flash attention] -> [GEMM O]
// All matmuls use v_wmma_f32_16x16x32_bf16 (f32 accumulate).
// GEMMs are LDS double-buffered (global loads overlap WMMA).
// Attention uses wave-private LDS only for the P C->A relayout, ordered with
// an inline `s_wait_dscnt 0` (no block barrier needed), and prefetches the
// next K tile / mask rows with global_prefetch_b8.
// ---------------------------------------------------------------------------

#define B_  8
#define SQ_ 2048
#define SK_ 2048
#define D_  512
#define H_  8
#define QP_ 512
#define VP_ 4096
#define DQH 64      // QP_/H_
#define DVH 512     // VP_/H_

typedef __bf16 bf16x16 __attribute__((ext_vector_type(16)));
typedef float  f32x8   __attribute__((ext_vector_type(8)));

union FragB16 {              // 16x bf16 == 8 dwords (A or B operand of wmma)
    unsigned int u[8];
    bf16x16      v;
};

__device__ __forceinline__ unsigned short f2bf(float f) {
    unsigned int x = __builtin_bit_cast(unsigned int, f);
    unsigned int r = (x + 0x7fffu + ((x >> 16) & 1u)) >> 16;   // RNE
    return (unsigned short)r;
}
__device__ __forceinline__ unsigned int pack2bf(float a, float b) {
    return (unsigned int)f2bf(a) | ((unsigned int)f2bf(b) << 16);
}

__device__ __forceinline__ f32x8 wmma_bf16(bf16x16 a, bf16x16 b, f32x8 c) {
    return __builtin_amdgcn_wmma_f32_16x16x32_bf16(
        false, a, false, b, (short)0, c, false, false);
}

// 16-bit A matrix 16x32, contiguous k order chosen here (not AMD's table order): VGPR v holds K pair 2v; lanes 16-31 shift K by 16.
__device__ __forceinline__ int amapK(int v, int half) {
    return 2 * v + half * 16;
}
// 16-bit B matrix 32x16, same contiguous k order as A (not AMD's table order): VGPR v holds K pair 2v; lanes 16-31 shift K by 16.
__device__ __forceinline__ int bmapK(int v, int half) {
    return 2 * v + half * 16;
}

// ---------------------------------------------------------------------------
// Generic GEMM: C[M,N] (bf16 or f32+bias) = A * B via bf16 WMMA.
// MODE 0: A f32 row-major (m,k)=Ap[m*ldA+k];    B=W f32 (k,n)=Bp[k*ldB+n]
// MODE 1: A bf16 row-major (m,k)=Ap[m*ldA+k];   B=W f32 (k,n)=Bp[k*ldB+n]
// MODE 2: A f32 transposed (m,k)=Ap[k*ldA+m];   B f32 (n,k)=Bp[n*ldB+k]
// Block tile 128x128, K-step 32, 256 threads = 8 waves (2x4), 8 wmma/step.
// LDS double-buffered: iteration i loads k-slice i+1 into registers, runs
// WMMA from buffer cur, writes registers to buffer cur^1, single barrier.
// ---------------------------------------------------------------------------
template <int MODE, bool OUT_F32>
__global__ __launch_bounds__(256) void gemm_kernel(
    const void* __restrict__ Ap, const void* __restrict__ Bp,
    void* __restrict__ Cp, const float* __restrict__ bias,
    int M, int N, int K, int ldA, int ldB)
{
    __shared__ __align__(16) unsigned short As[2][128 * 40];   // [m][k], pitch 40
    __shared__ __align__(16) unsigned short Bs[2][128 * 40];   // [n][k], pitch 40

    const int t    = threadIdx.x;
    const int m0   = blockIdx.x * 128;
    const int n0   = blockIdx.y * 128;
    const int w    = t >> 5, lane = t & 31;
    const int half = lane >> 4, li = lane & 15;
    const int wm   = w & 1, wn = w >> 1;

    // ---- staged-load helpers (global -> registers) ----
    float4 fa[4]; uint4 ua[2]; float4 fb[4];
    auto loadAB = [&](int kb) {
        if constexpr (MODE == 0) {
            const float* A = (const float*)Ap;
#pragma unroll
            for (int i = 0; i < 4; ++i) {
                int e = t + 256 * i, r = e >> 3, c4 = (e & 7) * 4;
                fa[i] = *(const float4*)(A + (size_t)(m0 + r) * ldA + kb + c4);
            }
        } else if constexpr (MODE == 1) {
            const unsigned short* A = (const unsigned short*)Ap;
#pragma unroll
            for (int i = 0; i < 2; ++i) {
                int e = t + 256 * i, r = e >> 2, c8 = (e & 3) * 8;
                ua[i] = *(const uint4*)(A + (size_t)(m0 + r) * ldA + kb + c8);
            }
        } else {
            const float* A = (const float*)Ap;
#pragma unroll
            for (int i = 0; i < 4; ++i) {
                int e = t + 256 * i, k = e >> 5, m4 = (e & 31) * 4;
                fa[i] = *(const float4*)(A + (size_t)(kb + k) * ldA + m0 + m4);
            }
        }
        if constexpr (MODE == 2) {
            const float* Bv = (const float*)Bp;
#pragma unroll
            for (int i = 0; i < 4; ++i) {
                int e = t + 256 * i, n = e >> 3, c4 = (e & 7) * 4;
                fb[i] = *(const float4*)(Bv + (size_t)(n0 + n) * ldB + kb + c4);
            }
        } else {
            const float* Wv = (const float*)Bp;
#pragma unroll
            for (int i = 0; i < 4; ++i) {
                int e = t + 256 * i, kr = e >> 5, n4 = (e & 31) * 4;
                fb[i] = *(const float4*)(Wv + (size_t)(kb + kr) * ldB + n0 + n4);
            }
        }
    };
    // ---- registers -> LDS buffer `buf` ----
    auto storeAB = [&](int buf) {
        if constexpr (MODE == 0) {
#pragma unroll
            for (int i = 0; i < 4; ++i) {
                int e = t + 256 * i, r = e >> 3, c4 = (e & 7) * 4;
                uint2 p; p.x = pack2bf(fa[i].x, fa[i].y); p.y = pack2bf(fa[i].z, fa[i].w);
                *(uint2*)&As[buf][r * 40 + c4] = p;
            }
        } else if constexpr (MODE == 1) {
#pragma unroll
            for (int i = 0; i < 2; ++i) {
                int e = t + 256 * i, r = e >> 2, c8 = (e & 3) * 8;
                *(uint4*)&As[buf][r * 40 + c8] = ua[i];
            }
        } else {
#pragma unroll
            for (int i = 0; i < 4; ++i) {
                int e = t + 256 * i, k = e >> 5, m4 = (e & 31) * 4;
                As[buf][(m4 + 0) * 40 + k] = f2bf(fa[i].x);
                As[buf][(m4 + 1) * 40 + k] = f2bf(fa[i].y);
                As[buf][(m4 + 2) * 40 + k] = f2bf(fa[i].z);
                As[buf][(m4 + 3) * 40 + k] = f2bf(fa[i].w);
            }
        }
        if constexpr (MODE == 2) {
#pragma unroll
            for (int i = 0; i < 4; ++i) {
                int e = t + 256 * i, n = e >> 3, c4 = (e & 7) * 4;
                uint2 p; p.x = pack2bf(fb[i].x, fb[i].y); p.y = pack2bf(fb[i].z, fb[i].w);
                *(uint2*)&Bs[buf][n * 40 + c4] = p;
            }
        } else {
#pragma unroll
            for (int i = 0; i < 4; ++i) {
                int e = t + 256 * i, kr = e >> 5, n4 = (e & 31) * 4;
                Bs[buf][(n4 + 0) * 40 + kr] = f2bf(fb[i].x);
                Bs[buf][(n4 + 1) * 40 + kr] = f2bf(fb[i].y);
                Bs[buf][(n4 + 2) * 40 + kr] = f2bf(fb[i].z);
                Bs[buf][(n4 + 3) * 40 + kr] = f2bf(fb[i].w);
            }
        }
    };

    f32x8 acc[4][2];
#pragma unroll
    for (int i = 0; i < 4; ++i)
#pragma unroll
        for (int j = 0; j < 2; ++j) acc[i][j] = (f32x8){0.f,0.f,0.f,0.f,0.f,0.f,0.f,0.f};

    // prologue: fill buffer 0
    loadAB(0);
    storeAB(0);
    __syncthreads();

    int cur = 0;
    for (int kb = 0; kb < K; kb += 32) {
        const bool nxt = (kb + 32) < K;
        if (nxt) loadAB(kb + 32);              // global loads overlap WMMA below

        FragB16 af[4], bfr[2];
#pragma unroll
        for (int ms = 0; ms < 4; ++ms) {
            int row = wm * 64 + ms * 16 + li;
#pragma unroll
            for (int v = 0; v < 8; ++v)
                af[ms].u[v] = *(const unsigned int*)&As[cur][row * 40 + amapK(v, half)];
        }
#pragma unroll
        for (int ns = 0; ns < 2; ++ns) {
            int nn = wn * 32 + ns * 16 + li;
#pragma unroll
            for (int v = 0; v < 8; ++v)
                bfr[ns].u[v] = *(const unsigned int*)&Bs[cur][nn * 40 + bmapK(v, half)];
        }
#pragma unroll
        for (int ms = 0; ms < 4; ++ms)
#pragma unroll
            for (int ns = 0; ns < 2; ++ns)
                acc[ms][ns] = wmma_bf16(af[ms].v, bfr[ns].v, acc[ms][ns]);

        if (nxt) storeAB(cur ^ 1);
        __syncthreads();
        cur ^= 1;
    }

    // store C: lane li = column n, VGPR r = row r (+8 for upper lane half)
#pragma unroll
    for (int ms = 0; ms < 4; ++ms)
#pragma unroll
        for (int ns = 0; ns < 2; ++ns)
#pragma unroll
            for (int r = 0; r < 8; ++r) {
                int row = m0 + wm * 64 + ms * 16 + r + half * 8;
                int col = n0 + wn * 32 + ns * 16 + li;
                float val = acc[ms][ns][r];
                if constexpr (OUT_F32)
                    ((float*)Cp)[(size_t)row * N + col] = val + (bias ? bias[col] : 0.f);
                else
                    ((unsigned short*)Cp)[(size_t)row * N + col] = f2bf(val);
            }
}

// ---------------------------------------------------------------------------
// Flash attention. Block = (b, h, 64 query rows), 256 threads = 8 waves:
// wave = (qi 0..3: 16-row q subtile) x (dj 0..1: 256-wide dv half).
// Q fragments hoisted. Per 32-wide K tile: mask loads issued first (latency
// hidden under S), S = Q*K^T with both nt fragments loaded before the wmmas,
// online softmax (16-lane shfl reductions), P relayout C->A through
// wave-private LDS ordered by s_wait_dscnt, then 16 wmma of P@V with a
// 2-deep software pipeline on the V^T fragments (contiguous b128 loads).
// ---------------------------------------------------------------------------
__global__ __launch_bounds__(256) void attn_flash_kernel(
    const unsigned short* __restrict__ Qb, const unsigned short* __restrict__ Kb,
    const unsigned short* __restrict__ Vt, const int* __restrict__ mask,
    unsigned short* __restrict__ Ctx)
{
    __shared__ __align__(16) unsigned short Pbuf[8 * 16 * 32]; // per-wave P tile

    const int bh = blockIdx.x;
    const int b  = bh / H_;
    const int h  = bh % H_;
    const int q0 = blockIdx.y * 64;

    const int t    = threadIdx.x;
    const int w    = t >> 5, lane = t & 31;
    const int half = lane >> 4, li = lane & 15;
    const int qi   = w & 3, dj = w >> 2;
    const int qr0  = q0 + qi * 16;

    f32x8 acc[16];
#pragma unroll
    for (int i = 0; i < 16; ++i) acc[i] = (f32x8){0.f,0.f,0.f,0.f,0.f,0.f,0.f,0.f};
    float mrow[8], lrow[8];
#pragma unroll
    for (int r = 0; r < 8; ++r) { mrow[r] = -3.0e38f; lrow[r] = 0.f; }

    // ---- hoist Q fragments (this wave's 16 rows x dq=64) ----
    FragB16 aq[2];
#pragma unroll
    for (int kk2 = 0; kk2 < 2; ++kk2) {
        size_t qbase = ((size_t)(b * SQ_ + qr0 + li)) * QP_ + h * DQH + kk2 * 32;
#pragma unroll
        for (int v = 0; v < 8; ++v)
            aq[kk2].u[v] = *(const unsigned int*)(Qb + qbase + amapK(v, half));
    }

    const size_t mbase0 = ((size_t)(b * SQ_ + qr0 + half * 8)) * SK_ + li;
    const size_t kbase0 = ((size_t)(b * SK_ + li)) * QP_ + h * DQH;

    for (int sk0 = 0; sk0 < SK_; sk0 += 32) {
        // ---- issue mask loads first: latency hides under S wmmas ----
        int mv[2][8];
#pragma unroll
        for (int nt = 0; nt < 2; ++nt)
#pragma unroll
            for (int r = 0; r < 8; ++r)
                mv[nt][r] = mask[mbase0 + (size_t)r * SK_ + sk0 + nt * 16];

        // ---- prefetch next K tile + mask rows (global_prefetch_b8) ----
        if (sk0 + 32 < SK_) {
            __builtin_prefetch(Kb + kbase0 + (size_t)(sk0 + 32) * QP_, 0, 1);
            __builtin_prefetch(Kb + kbase0 + (size_t)(sk0 + 48) * QP_, 0, 1);
            __builtin_prefetch(mask + mbase0 + sk0 + 32, 0, 1);
            __builtin_prefetch(mask + mbase0 + 4 * SK_ + sk0 + 32, 0, 1);
        }

        // ---- S = Q * K^T (16 rows x 32 sk cols) ----
        f32x8 s[2];
        s[0] = (f32x8){0.f,0.f,0.f,0.f,0.f,0.f,0.f,0.f};
        s[1] = s[0];
#pragma unroll
        for (int kk2 = 0; kk2 < 2; ++kk2) {
            FragB16 bk[2];                    // both nt fragments before wmmas
#pragma unroll
            for (int nt = 0; nt < 2; ++nt) {
                size_t kbase = kbase0 + (size_t)(sk0 + nt * 16) * QP_ + kk2 * 32;
#pragma unroll
                for (int v = 0; v < 8; ++v)
                    bk[nt].u[v] = *(const unsigned int*)(Kb + kbase + bmapK(v, half));
            }
#pragma unroll
            for (int nt = 0; nt < 2; ++nt)
                s[nt] = wmma_bf16(aq[kk2].v, bk[nt].v, s[nt]);
        }

        // ---- mask + scale (1/sqrt(D/H) = 1/8) ----
#pragma unroll
        for (int nt = 0; nt < 2; ++nt)
#pragma unroll
            for (int r = 0; r < 8; ++r) {
                float x = s[nt][r];
                x = (mv[nt][r] == 0) ? -1.0e9f : x;
                s[nt][r] = x * 0.125f;
            }

        // ---- online softmax update + P -> wave-private LDS ----
#pragma unroll
        for (int r = 0; r < 8; ++r) {
            float mx = fmaxf(s[0][r], s[1][r]);
#pragma unroll
            for (int off = 1; off < 16; off <<= 1)
                mx = fmaxf(mx, __shfl_xor(mx, off, 32));
            float mnew  = fmaxf(mrow[r], mx);
            float alpha = __expf(mrow[r] - mnew);
            float p0 = __expf(s[0][r] - mnew);
            float p1 = __expf(s[1][r] - mnew);
            float ls = p0 + p1;
#pragma unroll
            for (int off = 1; off < 16; off <<= 1)
                ls += __shfl_xor(ls, off, 32);
            lrow[r] = lrow[r] * alpha + ls;
            mrow[r] = mnew;
#pragma unroll
            for (int i = 0; i < 16; ++i) acc[i][r] *= alpha;
            int m = r + half * 8;
            Pbuf[w * 512 + m * 32 + li]      = f2bf(p0);
            Pbuf[w * 512 + m * 32 + 16 + li] = f2bf(p1);
        }
        // wave-private LDS round-trip: DS ops of one wave complete in order;
        // the wait + memory clobber pins both hardware and compiler ordering.
        asm volatile("s_wait_dscnt 0" ::: "memory");

        // ---- acc += P @ V : A-frag from LDS, pipelined B-frags from Vt ----
        FragB16 ap;
#pragma unroll
        for (int v = 0; v < 8; ++v)
            ap.u[v] = *(const unsigned int*)&Pbuf[w * 512 + li * 32 + amapK(v, half)];

        const size_t vbase0 = ((size_t)(h * DVH + dj * 256 + li)) * (B_ * SK_)
                            + b * SK_ + sk0;
        FragB16 bv[2];
#pragma unroll
        for (int v = 0; v < 8; ++v)
            bv[0].u[v] = *(const unsigned int*)(Vt + vbase0 + bmapK(v, half));
#pragma unroll
        for (int nt2 = 0; nt2 < 16; ++nt2) {
            if (nt2 < 15) {
                size_t vb = vbase0 + (size_t)(nt2 + 1) * 16 * (B_ * SK_);
#pragma unroll
                for (int v = 0; v < 8; ++v)
                    bv[(nt2 + 1) & 1].u[v] = *(const unsigned int*)(Vt + vb + bmapK(v, half));
            }
            acc[nt2] = wmma_bf16(ap.v, bv[nt2 & 1].v, acc[nt2]);
        }
    }

    // ---- finalize: /l, write ctx in the reference's (H,B,Sq,dv) flat order ----
#pragma unroll
    for (int nt2 = 0; nt2 < 16; ++nt2)
#pragma unroll
        for (int r = 0; r < 8; ++r) {
            int qrow = qr0 + r + half * 8;
            int dv   = dj * 256 + nt2 * 16 + li;
            float val = acc[nt2][r] / lrow[r];
            size_t idx = (((size_t)(h * B_ + b) * SQ_) + qrow) * (size_t)DVH + dv;
            Ctx[idx] = f2bf(val);
        }
}

// ---------------------------------------------------------------------------
extern "C" void kernel_launch(void* const* d_in, const int* in_sizes, int n_in,
                              void* d_out, int out_size, void* d_ws, size_t ws_size,
                              hipStream_t stream) {
    (void)in_sizes; (void)n_in; (void)out_size; (void)ws_size;
    const float* q_in = (const float*)d_in[0];
    const float* k_in = (const float*)d_in[1];
    const float* v_in = (const float*)d_in[2];
    const float* W_q  = (const float*)d_in[3];
    const float* W_k  = (const float*)d_in[4];
    const float* W_v  = (const float*)d_in[5];
    const float* W_o  = (const float*)d_in[6];
    const float* b_o  = (const float*)d_in[7];
    const int*   mask = (const int*)d_in[8];
    float*       out  = (float*)d_out;

    const size_t M = (size_t)B_ * SQ_;          // 16384
    unsigned short* Qb  = (unsigned short*)d_ws;
    unsigned short* Kb  = Qb + M * QP_;
    unsigned short* Vt  = Kb + M * QP_;         // transposed V: [VP_][M]
    unsigned short* Ctx = Vt + M * VP_;         // total ws use: 288 MB

    // Q/K projections (f32 in -> bf16 out, row-major)
    gemm_kernel<0, false><<<dim3(128, 4), 256, 0, stream>>>(
        q_in, W_q, Qb, nullptr, (int)M, QP_, D_, D_, QP_);
    gemm_kernel<0, false><<<dim3(128, 4), 256, 0, stream>>>(
        k_in, W_k, Kb, nullptr, (int)M, QP_, D_, D_, QP_);

    // V projection computed TRANSPOSED: Vt = W_v^T * v_in^T  (4096 x 16384)
    gemm_kernel<2, false><<<dim3(32, 128), 256, 0, stream>>>(
        W_v, v_in, Vt, nullptr, VP_, (int)M, D_, VP_, D_);

    // Flash attention -> ctx (bf16, already in the reference's buggy order)
    attn_flash_kernel<<<dim3(B_ * H_, SQ_ / 64), 256, 0, stream>>>(
        Qb, Kb, Vt, mask, Ctx);

    // Output projection (bf16 A, f32 out + bias)
    gemm_kernel<1, true><<<dim3(128, 4), 256, 0, stream>>>(
        Ctx, W_o, out, b_o, (int)M, D_, VP_, VP_, QP_);
}
